// IntentSpace_51187420234136
// MI455X (gfx1250) — matrix operand order repaired and checked by reading, not compiled, not run on hardware
//
#include <hip/hip_runtime.h>
#include <math.h>

typedef __attribute__((ext_vector_type(16))) __bf16 v16bf;
typedef __attribute__((ext_vector_type(8)))  float  v8f;

#define T_LEN   256
#define IN_DIM  2048
#define H_DIM   2048
#define B_DIM   8
#define SCAN_BLOCKS 16   // 16 blocks * 8 waves = 128 waves = 2048 rows / 16

union BfVec { uint4 q[2]; v16bf v; };

__device__ __forceinline__ unsigned short f32_to_bf16(float f) {
  unsigned int u = __float_as_uint(f);
  unsigned int r = u + 0x7FFFu + ((u >> 16) & 1u);   // round-to-nearest-even
  return (unsigned short)(r >> 16);
}

// ---------------------------------------------------------------------------
// Prep: reset grid-barrier counter (graph-replay safe), softmax the alpha row.
// ---------------------------------------------------------------------------
__global__ void prep_kernel(const float* __restrict__ alphas,
                            const int* __restrict__ alpha_int,
                            float* __restrict__ alpha,
                            unsigned int* __restrict__ counter) {
  if (threadIdx.x == 0) {
    *counter = 0u;
    int row = *alpha_int;
    float v[B_DIM];
    float mx = -1e30f;
    for (int j = 0; j < B_DIM; ++j) { v[j] = alphas[row * B_DIM + j]; mx = fmaxf(mx, v[j]); }
    float s = 0.f;
    for (int j = 0; j < B_DIM; ++j) { v[j] = expf(v[j] - mx); s += v[j]; }
    float inv = 1.f / s;
    for (int j = 0; j < B_DIM; ++j) alpha[j] = v[j] * inv;
  }
}

// ---------------------------------------------------------------------------
// fp32 -> bf16 grid-stride converter
// ---------------------------------------------------------------------------
__global__ void convert_bf16_kernel(const float* __restrict__ src,
                                    unsigned short* __restrict__ dst, int n) {
  for (int i = blockIdx.x * blockDim.x + threadIdx.x; i < n; i += gridDim.x * blockDim.x)
    dst[i] = f32_to_bf16(src[i]);
}

// ---------------------------------------------------------------------------
// u_bf16 = bf16( sum_b alpha[b] * W[b] )   (128 MB read, HBM-bound)
// ---------------------------------------------------------------------------
__global__ void mix_kernel(const float* __restrict__ W,
                           const float* __restrict__ alpha,
                           unsigned short* __restrict__ ub) {
  const int n = H_DIM * H_DIM;
  float al[B_DIM];
#pragma unroll
  for (int b = 0; b < B_DIM; ++b) al[b] = alpha[b];
  for (int i = blockIdx.x * blockDim.x + threadIdx.x; i < n; i += gridDim.x * blockDim.x) {
    float s = 0.f;
#pragma unroll
    for (int b = 0; b < B_DIM; ++b) s = fmaf(al[b], W[b * n + i], s);
    ub[i] = f32_to_bf16(s);
  }
}

// ---------------------------------------------------------------------------
// igates = x @ v.T  via v_wmma_f32_16x16x32_bf16.
// One wave per 16x16 tile. B = v.T in column-major == rows of v row-major,
// so both A and B fragments are two B128 loads per lane per K=32 chunk.
// A layout (16-bit, 16x32):  lanes 0-15 M=l16, halves0-7 = K k0+hi*8..,
//                            halves8-15 = K +16.  (hi = lane>=16 -> K base +8)
// B layout (16-bit, 32x16):  lane -> N=l16, same K order as A:
//                            halves0-7 = K k0+hi*8.., halves8-15 = K +16.
// ---------------------------------------------------------------------------
__global__ void igates_gemm_kernel(const unsigned short* __restrict__ xb,
                                   const unsigned short* __restrict__ vb,
                                   float* __restrict__ ig) {
  const int wave = blockIdx.x * 8 + (threadIdx.x >> 5);  // 2048 waves
  const int lane = threadIdx.x & 31;
  const int l16  = lane & 15;
  const int hi   = lane >> 4;
  const int m0 = (wave >> 7) * 16;    // 16 row tiles (T=256)
  const int n0 = (wave & 127) * 16;   // 128 col tiles (H=2048)

  v8f acc = {0.f, 0.f, 0.f, 0.f, 0.f, 0.f, 0.f, 0.f};
  const unsigned short* arow = xb + (m0 + l16) * IN_DIM + hi * 8;
  const unsigned short* brow = vb + (n0 + l16) * IN_DIM + hi * 8;

  for (int k0 = 0; k0 < IN_DIM; k0 += 32) {
    BfVec a, b;
    a.q[0] = *(const uint4*)(arow + k0);
    a.q[1] = *(const uint4*)(arow + k0 + 16);
    b.q[0] = *(const uint4*)(brow + k0);
    b.q[1] = *(const uint4*)(brow + k0 + 16);
    acc = __builtin_amdgcn_wmma_f32_16x16x32_bf16(false, a.v, false, b.v,
                                                  (short)0, acc, false, false);
  }
  // D layout: lane -> N=l16; VGPR r -> M = r + 8*hi
  float* out = ig + (m0 + hi * 8) * H_DIM + n0 + l16;
#pragma unroll
  for (int r = 0; r < 8; ++r) out[r * H_DIM] = acc[r];
}

// ---------------------------------------------------------------------------
// Grid-wide barrier: monotonic counter, release/acquire via threadfence.
// ---------------------------------------------------------------------------
__device__ __forceinline__ void grid_barrier(unsigned int* counter, unsigned int target) {
  __syncthreads();
  __threadfence();                       // release our h writes
  if (threadIdx.x == 0) {
    __hip_atomic_fetch_add(counter, 1u, __ATOMIC_RELEASE, __HIP_MEMORY_SCOPE_AGENT);
    while (__hip_atomic_load(counter, __ATOMIC_ACQUIRE, __HIP_MEMORY_SCOPE_AGENT) < target)
      __builtin_amdgcn_s_sleep(1);
  }
  __syncthreads();
  __threadfence();                       // acquire everyone else's h writes
}

// ---------------------------------------------------------------------------
// Persistent scan: 256 steps of h = tanh(ig_t + u@h + bias).
// Each wave owns a 16-row M-tile of u for the whole scan; per step the matvec
// is WMMA with h packed in column N=0 of B (lanes 0 & 16 carry it). A-stream
// (u, bf16, 8 MB) lives in L2; the wasted B/D columns are free — step is
// bandwidth-bound. Double-buffered bf16 h, device barrier between steps.
// ---------------------------------------------------------------------------
__global__ void scan_kernel(const unsigned short* __restrict__ ub,
                            const float* __restrict__ ig,
                            const float* __restrict__ bias,
                            float* __restrict__ hf,
                            unsigned short* __restrict__ hbf,   // [2][H_DIM]
                            unsigned int* __restrict__ counter) {
  const int wave = blockIdx.x * 8 + (threadIdx.x >> 5);  // 128 waves
  const int lane = threadIdx.x & 31;
  const int l16  = lane & 15;
  const int hi   = lane >> 4;
  const int m0   = wave * 16;
  unsigned int phase = 0;

  // init h = 0 (buffer 0): lanes 0 & 16 own 8 rows each of this tile
  if (l16 == 0) {
#pragma unroll
    for (int r = 0; r < 8; ++r) {
      int row = m0 + hi * 8 + r;
      hf[row]  = 0.f;
      hbf[row] = 0;
    }
  }
  ++phase; grid_barrier(counter, SCAN_BLOCKS * phase);

  const unsigned short* arow = ub + (m0 + l16) * H_DIM + hi * 8;

  for (int t = 0; t < T_LEN; ++t) {
    const unsigned short* hsrc = hbf + (t & 1) * H_DIM;
    unsigned short*       hdst = hbf + ((t + 1) & 1) * H_DIM;

    v8f acc = {0.f, 0.f, 0.f, 0.f, 0.f, 0.f, 0.f, 0.f};
    for (int k0 = 0; k0 < H_DIM; k0 += 32) {
      BfVec a, b = {};
      a.q[0] = *(const uint4*)(arow + k0);
      a.q[1] = *(const uint4*)(arow + k0 + 16);
      if (l16 == 0) {                      // column N=0 of B holds h
        b.q[0] = *(const uint4*)(hsrc + k0 + hi * 8);
        b.q[1] = *(const uint4*)(hsrc + k0 + hi * 8 + 16);
      }
      acc = __builtin_amdgcn_wmma_f32_16x16x32_bf16(false, a.v, false, b.v,
                                                    (short)0, acc, false, false);
    }
    if (l16 == 0) {                        // D column 0: lane0 M=r, lane16 M=8+r
#pragma unroll
      for (int r = 0; r < 8; ++r) {
        int row = m0 + hi * 8 + r;
        float y = tanhf(ig[t * H_DIM + row] + acc[r] + bias[row]);
        hf[row]  = y;
        hdst[row] = f32_to_bf16(y);
      }
    }
    ++phase; grid_barrier(counter, SCAN_BLOCKS * phase);
  }
}

// ---------------------------------------------------------------------------
// out = sigmoid(h @ fc_w.T) : one wave per output row (fc_b not added).
// ---------------------------------------------------------------------------
__global__ void fc_kernel(const float* __restrict__ hf,
                          const float* __restrict__ fw,
                          const float* __restrict__ fb,
                          float* __restrict__ out) {
  const int row  = blockIdx.x * 8 + (threadIdx.x >> 5);
  const int lane = threadIdx.x & 31;
  const float* wr = fw + row * H_DIM;
  float s = 0.f;
  for (int k = lane; k < H_DIM; k += 32) s = fmaf(wr[k], hf[k], s);
#pragma unroll
  for (int off = 16; off; off >>= 1) s += __shfl_xor(s, off);
  if (lane == 0) out[row] = 1.f / (1.f + expf(-s));
}

// ---------------------------------------------------------------------------
// Workspace layout (bytes); total ~19 MB.
// ---------------------------------------------------------------------------
#define OFF_ALPHA 0u
#define OFF_CNT   64u
#define OFF_HF    256u                                   // 2048 f32
#define OFF_HBF   (OFF_HF + H_DIM * 4u)                  // 2*2048 bf16
#define OFF_IG    (OFF_HBF + 2u * H_DIM * 2u)            // 256*2048 f32
#define OFF_XBF   (OFF_IG + (unsigned)T_LEN * H_DIM * 4u)    // 256*2048 bf16
#define OFF_VBF   (OFF_XBF + (unsigned)T_LEN * IN_DIM * 2u)  // 2048*2048 bf16
#define OFF_UBF   (OFF_VBF + (unsigned)H_DIM * IN_DIM * 2u)  // 2048*2048 bf16

extern "C" void kernel_launch(void* const* d_in, const int* in_sizes, int n_in,
                              void* d_out, int out_size, void* d_ws, size_t ws_size,
                              hipStream_t stream) {
  const float* x      = (const float*)d_in[0];   // [T, IN]
  const float* W      = (const float*)d_in[1];   // [B, H, H]
  const float* alphas = (const float*)d_in[2];   // [B, B]
  const float* bias   = (const float*)d_in[3];   // [H]
  const float* v      = (const float*)d_in[4];   // [H, IN]
  const float* fc_w   = (const float*)d_in[5];   // [H, H]
  const float* fc_b   = (const float*)d_in[6];   // [H]
  const int*   a_int  = (const int*)d_in[7];

  char* ws = (char*)d_ws;
  float*          alpha  = (float*)(ws + OFF_ALPHA);
  unsigned int*   cnt    = (unsigned int*)(ws + OFF_CNT);
  float*          hf     = (float*)(ws + OFF_HF);
  unsigned short* hbf    = (unsigned short*)(ws + OFF_HBF);
  float*          ig     = (float*)(ws + OFF_IG);
  unsigned short* xbf    = (unsigned short*)(ws + OFF_XBF);
  unsigned short* vbf    = (unsigned short*)(ws + OFF_VBF);
  unsigned short* ubf    = (unsigned short*)(ws + OFF_UBF);
  float*          out    = (float*)d_out;

  prep_kernel<<<1, 32, 0, stream>>>(alphas, a_int, alpha, cnt);
  convert_bf16_kernel<<<512, 256, 0, stream>>>(x, xbf, T_LEN * IN_DIM);
  convert_bf16_kernel<<<1024, 256, 0, stream>>>(v, vbf, H_DIM * IN_DIM);
  mix_kernel<<<1024, 256, 0, stream>>>(W, alpha, ubf);
  igates_gemm_kernel<<<256, 256, 0, stream>>>(xbf, vbf, ig);
  scan_kernel<<<SCAN_BLOCKS, 256, 0, stream>>>(ubf, ig, bias, hf, hbf, cnt);
  fc_kernel<<<256, 256, 0, stream>>>(hf, fc_w, fc_b, out);
}
